// PoolingFineNetWithAppearance_52948356825726
// MI455X (gfx1250) — compile-verified, matrix operand order checked by reading, not run on hardware
//
#include <hip/hip_runtime.h>
#include <cstdint>
#include <cstddef>
#include <cstring>
#include <cfloat>

// ---------------------------------------------------------------------------
// Problem constants (from the reference)
// ---------------------------------------------------------------------------
#define NN   20000
#define EE   320000
#define K1N  10000      // N/2
#define K2N  5000       // (K1+1)/2
#define NFW  64

typedef __attribute__((ext_vector_type(16))) _Float16 v16h;
typedef __attribute__((ext_vector_type(8)))  _Float16 v8h;
typedef __attribute__((ext_vector_type(8)))  float    v8f;

#define FATOMIC_ADD(p, v) unsafeAtomicAdd((p), (v))

// ---- gfx1250 async global->LDS path (guarded; falls back cleanly) ----------
#define AS_LOCAL  __attribute__((address_space(3)))
typedef __attribute__((__vector_size__(4 * sizeof(int)))) int vsi4;  // builtin's param type
#if __has_builtin(__builtin_amdgcn_global_load_async_to_lds_b128)
#define HAVE_ASYNC_LDS 1
#endif

__device__ __forceinline__ void async_wait0() {
#if defined(HAVE_ASYNC_LDS)
#if __has_builtin(__builtin_amdgcn_s_wait_asynccnt)
  __builtin_amdgcn_s_wait_asynccnt(0);
#else
  asm volatile("s_wait_asynccnt 0x0" ::: "memory");
#endif
#endif
}

// ---------------------------------------------------------------------------
// Quaternion helpers
// ---------------------------------------------------------------------------
__device__ __forceinline__ void qmul_(const float* a, const float* b, float* o) {
  o[0] = a[0]*b[0] - a[1]*b[1] - a[2]*b[2] - a[3]*b[3];
  o[1] = a[0]*b[1] + a[1]*b[0] + a[2]*b[3] - a[3]*b[2];
  o[2] = a[0]*b[2] - a[1]*b[3] + a[2]*b[0] + a[3]*b[1];
  o[3] = a[0]*b[3] + a[1]*b[2] - a[2]*b[1] + a[3]*b[0];
}
__device__ __forceinline__ void qinv_(const float* a, float* o) {
  o[0] = a[0]; o[1] = -a[1]; o[2] = -a[2]; o[3] = -a[3];
}
__device__ __forceinline__ void qnorm_(float* a) {
  float n = sqrtf(a[0]*a[0] + a[1]*a[1] + a[2]*a[2] + a[3]*a[3]);
  n = fmaxf(n, 1e-12f);
  a[0] /= n; a[1] /= n; a[2] /= n; a[3] /= n;
}

// ordered-int key for float atomic min/max (monotonic under signed compare)
__device__ __forceinline__ int   fkey(float f)  { int i = __float_as_int(f); return i >= 0 ? i : (i ^ 0x7fffffff); }
__device__ __forceinline__ float funkey(int k)  { int i = k >= 0 ? k : (k ^ 0x7fffffff); return __int_as_float(i); }

// host-side copy of fkey (bit-cast without device intrinsics)
static inline int fkey_host(float f) {
  int i; std::memcpy(&i, &f, sizeof(int));
  return i >= 0 ? i : (i ^ 0x7fffffff);
}

// ---------------------------------------------------------------------------
// Weight pre-conversion: w (din x dout, f32, row-major) -> wt (dout x din_p,
// f16, N-major, zero-padded K). Done once per layer; panels are then staged
// into LDS with raw 16-byte (or async) copies inside the WMMA kernel.
// ---------------------------------------------------------------------------
__global__ void wcvt_kernel(_Float16* wt, const float* w, int din, int din_p, int dout) {
  int i = blockIdx.x * blockDim.x + threadIdx.x;
  if (i >= dout * din_p) return;
  int n = i / din_p, k = i % din_p;
  wt[i] = (_Float16)((k < din) ? w[(size_t)k * dout + n] : 0.0f);
}

// ---------------------------------------------------------------------------
// Fused edge-conv:  e = ReLU(concat(x[src],x[dst],eattr) @ W1 + b1) @ W2 + b2
//                   s[dst] += e * mask   (atomic segment sum)
// Tile: 128 edges / block, 8 waves, each wave = one 16-edge strip, WMMA f16.
// ---------------------------------------------------------------------------
struct EConvIn {
  const float* xa; const float* xb; int wa, wb, nin;   // node concat segments
  const float* ea; const float* eb; int wea, web;      // edge concat segments
  const int* src; const int* dst; const float* mask;   // mask==null -> 1.0
  const _Float16* w1t;  // dout x din_p, N-major f16
  const _Float16* w2t;  // dout x dout,  N-major f16
  const float* b1; const float* b2;
  float* e_out; float* s_out;
  int E; int din; int din_p;
};

template <int DOUT>
__global__ void __launch_bounds__(256) econv_fused(EConvIn A) {
  constexpr int NT = DOUT / 16;
  __shared__ __align__(16) _Float16 sA[128 * 32];       // A staging (128 edges x 32 K)
  __shared__ __align__(16) _Float16 sT[DOUT * 32];      // W panel, N-major: [n][k]
  __shared__ __align__(16) _Float16 sH[128 * DOUT];     // hidden activations (f16)
  __shared__ int   sS[128];
  __shared__ int   sD[128];
  __shared__ float sM[128];

  const int tid = threadIdx.x;
  const int tb  = blockIdx.x * 128;

  for (int i = tid; i < 128; i += 256) {
    int e = tb + i;
    sS[i] = (e < A.E) ? A.src[e] : 0;
    sD[i] = (e < A.E) ? A.dst[e] : 0;
    sM[i] = (e < A.E) ? (A.mask ? A.mask[e] : 1.0f) : 0.0f;
  }
  __syncthreads();

  const int lane = tid & 31;
  const int wv   = tid >> 5;
  const int r16  = lane & 15;
  const int hh   = lane >> 4;     // lane half: selects K sub-chunk per ISA layout
  const int nin2 = 2 * A.nin;

  v8f zero{};
  v8f acc[NT];
#pragma unroll
  for (int t = 0; t < NT; ++t) acc[t] = zero;

  // ---------------- GEMM1 over K = din (padded to 32) ----------------
  const int kst = (A.din + 31) >> 5;
  for (int ks = 0; ks < kst; ++ks) {
    const int k0 = ks * 32;
    // stage W1 K-panel first (async if available, overlaps with A gather)
    for (int idx = tid; idx < DOUT * 4; idx += 256) {
      int n = idx >> 2, ch = idx & 3;
#if defined(HAVE_ASYNC_LDS)
      __builtin_amdgcn_global_load_async_to_lds_b128(
          (vsi4*)&A.w1t[(size_t)n * A.din_p + k0 + ch * 8],
          (AS_LOCAL vsi4*)&sT[n * 32 + ch * 8], 0, 0);
#else
      *(v8h*)&sT[n * 32 + ch * 8] =
          *(const v8h*)&A.w1t[(size_t)n * A.din_p + k0 + ch * 8];
#endif
    }
    // stage A: coalesced gather+convert (32 lanes span one 128B row segment)
    for (int idx = tid; idx < 128 * 32; idx += 256) {
      int m = idx >> 5, kk = idx & 31, c = k0 + kk;
      float v = 0.0f;
      int e = tb + m;
      if (e < A.E && c < A.din) {
        if (c < nin2) {
          // node rows: small arrays, L2-resident gather
          int node = (c < A.nin) ? sS[m] : sD[m];
          int cc   = (c < A.nin) ? c : (c - A.nin);
          v = (cc < A.wa) ? A.xa[(size_t)node * A.wa + cc]
                          : A.xb[(size_t)node * A.wb + (cc - A.wa)];
        } else {
          // edge rows: the true HBM stream -> prefetch next K-step's line
          int ce = c - nin2;
          const float* p = (ce < A.wea) ? &A.ea[(size_t)e * A.wea + ce]
                                        : &A.eb[(size_t)e * A.web + (ce - A.wea)];
          v = *p;
          __builtin_prefetch(p + 32, 0, 1);   // +128B = next iteration's column block
        }
      }
      sA[m * 32 + kk] = (_Float16)v;
    }
    async_wait0();
    __syncthreads();

    // load A tile (16x32 f16, per ISA VGPR layout)
    const _Float16* pa = &sA[(wv * 16 + r16) * 32 + hh * 8];
    v8h a0 = *(const v8h*)pa;
    v8h a1 = *(const v8h*)(pa + 16);
    v16h av;
#pragma unroll
    for (int i = 0; i < 8; ++i) { av[i] = a0[i]; av[8 + i] = a1[i]; }

#pragma unroll
    for (int t = 0; t < NT; ++t) {
      const _Float16* pb = &sT[(t * 16 + r16) * 32 + hh * 8];
      v8h b0 = *(const v8h*)pb;
      v8h b1 = *(const v8h*)(pb + 16);
      v16h bv;
#pragma unroll
      for (int i = 0; i < 8; ++i) { bv[i] = b0[i]; bv[8 + i] = b1[i]; }
      acc[t] = __builtin_amdgcn_wmma_f32_16x16x32_f16(
          false, av, false, bv, (short)0, acc[t], false, false);
    }
    __syncthreads();
  }

  // epilogue 1: bias + ReLU -> sH (f16)
#pragma unroll
  for (int t = 0; t < NT; ++t) {
    int n = t * 16 + r16;
    float bn = A.b1[n];
#pragma unroll
    for (int rr = 0; rr < 8; ++rr) {
      int m = wv * 16 + rr + hh * 8;
      float v = acc[t][rr] + bn;
      sH[m * DOUT + n] = (_Float16)(v > 0.0f ? v : 0.0f);
    }
  }
  __syncthreads();

  // ---------------- GEMM2: e = h @ W2, K = DOUT ----------------
#pragma unroll
  for (int t = 0; t < NT; ++t) acc[t] = zero;
  for (int ks = 0; ks < DOUT / 32; ++ks) {
    const int k0 = ks * 32;
    for (int idx = tid; idx < DOUT * 4; idx += 256) {
      int n = idx >> 2, ch = idx & 3;
#if defined(HAVE_ASYNC_LDS)
      __builtin_amdgcn_global_load_async_to_lds_b128(
          (vsi4*)&A.w2t[(size_t)n * DOUT + k0 + ch * 8],
          (AS_LOCAL vsi4*)&sT[n * 32 + ch * 8], 0, 0);
#else
      *(v8h*)&sT[n * 32 + ch * 8] =
          *(const v8h*)&A.w2t[(size_t)n * DOUT + k0 + ch * 8];
#endif
    }
    async_wait0();
    __syncthreads();

    const _Float16* pa = &sH[(wv * 16 + r16) * DOUT + k0 + hh * 8];
    v8h a0 = *(const v8h*)pa;
    v8h a1 = *(const v8h*)(pa + 16);
    v16h av;
#pragma unroll
    for (int i = 0; i < 8; ++i) { av[i] = a0[i]; av[8 + i] = a1[i]; }

#pragma unroll
    for (int t = 0; t < NT; ++t) {
      const _Float16* pb = &sT[(t * 16 + r16) * 32 + hh * 8];
      v8h b0 = *(const v8h*)pb;
      v8h b1 = *(const v8h*)(pb + 16);
      v16h bv;
#pragma unroll
      for (int i = 0; i < 8; ++i) { bv[i] = b0[i]; bv[8 + i] = b1[i]; }
      acc[t] = __builtin_amdgcn_wmma_f32_16x16x32_f16(
          false, av, false, bv, (short)0, acc[t], false, false);
    }
    __syncthreads();
  }

  // epilogue 2: bias, stream e_out, atomic segment-sum into s_out
#pragma unroll
  for (int t = 0; t < NT; ++t) {
    int n = t * 16 + r16;
    float bn = A.b2[n];
#pragma unroll
    for (int rr = 0; rr < 8; ++rr) {
      int m = wv * 16 + rr + hh * 8;
      int e = tb + m;
      if (e < A.E) {
        float v = acc[t][rr] + bn;
        A.e_out[(size_t)e * DOUT + n] = v;
        float mm = sM[m];
        if (mm != 0.0f)
          FATOMIC_ADD(&A.s_out[(size_t)sD[m] * DOUT + n], v * mm);
      }
    }
  }
}

// ---------------------------------------------------------------------------
// Small supporting kernels
// ---------------------------------------------------------------------------
__global__ void eam_kernel(float* eam, const float* x, const float* eattr,
                           const int* src, const int* dst, int E) {
  int j = blockIdx.x * blockDim.x + threadIdx.x;
  if (j >= E) return;
  float qd[4], qi[4], e[4], t[4], u[4];
  const float* pd = x + (size_t)dst[j] * 4;
  const float* ps = x + (size_t)src[j] * 4;
  qd[0]=pd[0]; qd[1]=pd[1]; qd[2]=pd[2]; qd[3]=pd[3];
  qinv_(qd, qi);
  e[0]=eattr[(size_t)j*4+0]; e[1]=eattr[(size_t)j*4+1];
  e[2]=eattr[(size_t)j*4+2]; e[3]=eattr[(size_t)j*4+3];
  qmul_(qi, e, t);
  float qs[4] = {ps[0], ps[1], ps[2], ps[3]};
  qmul_(t, qs, u);
  qnorm_(u);
  eam[(size_t)j*4+0]=u[0]; eam[(size_t)j*4+1]=u[1];
  eam[(size_t)j*4+2]=u[2]; eam[(size_t)j*4+3]=u[3];
}

__global__ void seg_count_kernel(float* c, const int* dst, const float* mask, int E) {
  int j = blockIdx.x * blockDim.x + threadIdx.x;
  if (j >= E) return;
  float m = mask ? mask[j] : 1.0f;
  if (m != 0.0f) FATOMIC_ADD(&c[dst[j]], m);
}

__global__ void node_fin_kernel(float* x, const float* s, const float* c, int n, int w) {
  int i = blockIdx.x * blockDim.x + threadIdx.x;
  if (i >= n * w) return;
  float cc = fmaxf(c[i / w], 1.0f);
  float v  = s[i] / cc;
  x[i] = v > 0.0f ? v : 0.0f;
}

__global__ void relu_kernel(float* p, size_t n) {
  size_t i = (size_t)blockIdx.x * blockDim.x + threadIdx.x;
  if (i < n) { float v = p[i]; p[i] = v > 0.0f ? v : 0.0f; }
}

__global__ void fill_int_kernel(int* p, int v, int n) {
  int i = blockIdx.x * blockDim.x + threadIdx.x;
  if (i < n) p[i] = v;
}

// ---- GAT score ----
__global__ void gat_h_kernel(float* h, const float* x, const float* w, int n, int d) {
  int i = blockIdx.x * blockDim.x + threadIdx.x;
  if (i >= n) return;
  float s = 0.0f;
  for (int f = 0; f < d; ++f) s += x[(size_t)i * d + f] * w[f];
  h[i] = s;
}
__device__ __forceinline__ float leaky02(float v) { return v > 0.0f ? v : 0.2f * v; }

__global__ void gat_max_edges_kernel(int* mx, const float* h, const int* src, const int* dst,
                                     const float* mask, const float* asrc, const float* adst, int E) {
  int j = blockIdx.x * blockDim.x + threadIdx.x;
  if (j >= E) return;
  float m = mask ? mask[j] : 1.0f;
  if (m <= 0.0f) return;
  float l = leaky02(h[src[j]] * (*asrc) + h[dst[j]] * (*adst));
  atomicMax(&mx[dst[j]], fkey(l));
}
__global__ void gat_max_self_kernel(int* mx, const float* h,
                                    const float* asrc, const float* adst, int n) {
  int i = blockIdx.x * blockDim.x + threadIdx.x;
  if (i >= n) return;
  float l = leaky02(h[i] * (*asrc) + h[i] * (*adst));
  atomicMax(&mx[i], fkey(l));
}
__global__ void gat_acc_edges_kernel(float* den, float* num, const int* mx, const float* h,
                                     const int* src, const int* dst, const float* mask,
                                     const float* asrc, const float* adst, int E) {
  int j = blockIdx.x * blockDim.x + threadIdx.x;
  if (j >= E) return;
  float m = mask ? mask[j] : 1.0f;
  if (m <= 0.0f) return;
  int d = dst[j];
  float l  = leaky02(h[src[j]] * (*asrc) + h[d] * (*adst));
  float ex = __expf(l - funkey(mx[d]));
  FATOMIC_ADD(&den[d], ex);
  FATOMIC_ADD(&num[d], ex * h[src[j]]);
}
__global__ void gat_acc_self_kernel(float* den, float* num, const int* mx, const float* h,
                                    const float* asrc, const float* adst, int n) {
  int i = blockIdx.x * blockDim.x + threadIdx.x;
  if (i >= n) return;
  float l  = leaky02(h[i] * (*asrc) + h[i] * (*adst));
  float ex = __expf(l - funkey(mx[i]));
  FATOMIC_ADD(&den[i], ex);
  FATOMIC_ADD(&num[i], ex * h[i]);
}
__global__ void gat_score_kernel(float* score, const float* num, const float* den,
                                 const float* b, int n) {
  int i = blockIdx.x * blockDim.x + threadIdx.x;
  if (i >= n) return;
  score[i] = num[i] / fmaxf(den[i], 1e-16f) + (*b);
}

// ---- top-k via histogram threshold selection ----
__global__ void minmax_init_kernel(int* mm) {
  mm[0] = fkey(FLT_MAX); mm[1] = fkey(-FLT_MAX);
}
__global__ void minmax_kernel(int* mm, const float* s, int n) {
  int i = blockIdx.x * blockDim.x + threadIdx.x;
  if (i >= n) return;
  int k = fkey(s[i]);
  atomicMin(&mm[0], k);
  atomicMax(&mm[1], k);
}
__device__ __forceinline__ int score_bin(float v, float mn, float rng) {
  int b = (rng > 0.0f) ? (int)((v - mn) * (1024.0f / rng)) : 0;
  return b < 0 ? 0 : (b > 1023 ? 1023 : b);
}
__global__ void hist_kernel(int* hist, const int* mm, const float* s, int n) {
  int i = blockIdx.x * blockDim.x + threadIdx.x;
  if (i >= n) return;
  float mn = funkey(mm[0]), rng = funkey(mm[1]) - mn;
  atomicAdd(&hist[score_bin(s[i], mn, rng)], 1);
}
__global__ void threshold_kernel(int* thr, const int* hist, int k) {
  int cum = 0, bsel = 0;
  for (int b = 1023; b >= 0; --b) {
    if (cum + hist[b] >= k) { bsel = b; break; }
    cum += hist[b];
  }
  thr[0] = bsel; thr[1] = cum;   // cum = count strictly above bsel
}
__global__ void select_kernel(int* perm, float* vals, int* remap, int* ctr,
                              const int* thr, const int* mm, const float* s, int n, int k) {
  int i = blockIdx.x * blockDim.x + threadIdx.x;
  if (i >= n) return;
  float mn = funkey(mm[0]), rng = funkey(mm[1]) - mn;
  int b = score_bin(s[i], mn, rng);
  int bsel = thr[0], cnt_hi = thr[1];
  int pos = -1;
  if (b > bsel) {
    pos = atomicAdd(&ctr[0], 1);
  } else if (b == bsel) {
    int q = atomicAdd(&ctr[1], 1);
    if (q < k - cnt_hi) pos = cnt_hi + q;
  }
  if (pos >= 0) { perm[pos] = i; vals[pos] = s[i]; remap[i] = pos; }
}
__global__ void pool_rows_kernel(float* xp, const float* x, const int* perm,
                                 const float* vals, int k, int w) {
  int i = blockIdx.x * blockDim.x + threadIdx.x;
  if (i >= k * w) return;
  int row = i / w, col = i % w;
  xp[i] = x[(size_t)perm[row] * w + col] * tanhf(vals[row]);
}
__global__ void remap_edges_kernel(int* ns, int* nd, float* nm, const int* src, const int* dst,
                                   const float* mask, const int* remap, int E) {
  int j = blockIdx.x * blockDim.x + threadIdx.x;
  if (j >= E) return;
  int rs = remap[src[j]], rd = remap[dst[j]];
  float m = (mask ? mask[j] : 1.0f) * (rs >= 0 ? 1.0f : 0.0f) * (rd >= 0 ? 1.0f : 0.0f);
  ns[j] = rs > 0 ? rs : 0;
  nd[j] = rd > 0 ? rd : 0;
  nm[j] = m;
}
__global__ void rows_add_kernel(float* dst, const float* add, const int* perm, int k, int w) {
  int i = blockIdx.x * blockDim.x + threadIdx.x;
  if (i >= k * w) return;
  int row = i / w, col = i % w;
  dst[(size_t)perm[row] * w + col] += add[i];
}

// ---- heads + loss ----
__global__ void lin1_final_kernel(float* xout, const float* x5, const float* w,
                                  const float* b, const float* x_org, int n) {
  int i = blockIdx.x * blockDim.x + threadIdx.x;
  if (i >= n) return;
  float p[4];
  for (int o = 0; o < 4; ++o) {
    float s = b[o];
    for (int f = 0; f < NFW; ++f) s += x5[(size_t)i * NFW + f] * w[f * 4 + o];
    p[o] = s;
  }
  float q[4];
  const float* xo = x_org + (size_t)i * 4;
  float xoq[4] = {xo[0], xo[1], xo[2], xo[3]};
  qmul_(p, xoq, q);
  qnorm_(q);
  xout[(size_t)i*4+0]=q[0]; xout[(size_t)i*4+1]=q[1];
  xout[(size_t)i*4+2]=q[2]; xout[(size_t)i*4+3]=q[3];
}
__global__ void lin2_kernel(float* out, const float* e5, const float* w,
                            const float* b, int E) {
  int j = blockIdx.x * blockDim.x + threadIdx.x;
  if (j >= E) return;
  float s = b[0];
  for (int f = 0; f < NFW; ++f) s += e5[(size_t)j * NFW + f] * w[f];
  out[j] = s;
}
__global__ void loss_kernel(float* loss, const float* xfin, const float* gt,
                            const int* src, const int* dst, const float* beta, int E) {
  int j = blockIdx.x * blockDim.x + threadIdx.x;
  if (j >= E) return;
  int sj = src[j], dj = dst[j];
  float gs[4] = {gt[(size_t)sj*4+0], gt[(size_t)sj*4+1], gt[(size_t)sj*4+2], gt[(size_t)sj*4+3]};
  float gd[4] = {gt[(size_t)dj*4+0], gt[(size_t)dj*4+1], gt[(size_t)dj*4+2], gt[(size_t)dj*4+3]};
  float xs[4] = {xfin[(size_t)sj*4+0], xfin[(size_t)sj*4+1], xfin[(size_t)sj*4+2], xfin[(size_t)sj*4+3]};
  float xd[4] = {xfin[(size_t)dj*4+0], xfin[(size_t)dj*4+1], xfin[(size_t)dj*4+2], xfin[(size_t)dj*4+3]};
  float gsi[4], xsi[4], rg[4], rp[4], rgi[4], q[4];
  qinv_(gs, gsi); qmul_(gd, gsi, rg);
  qinv_(xs, xsi); qmul_(xd, xsi, rp);
  qinv_(rg, rgi); qmul_(rgi, rp, q);
  qnorm_(q);
  float be = beta[0];
  float tgt[4] = {1.0f, 0.0f, 0.0f, 0.0f};
  float acc = 0.0f;
  for (int c = 0; c < 4; ++c) {
    float d = fabsf(q[c] - tgt[c]);
    acc += (d < be) ? 0.5f * d * d / be : d - 0.5f * be;
  }
  FATOMIC_ADD(loss, acc * (1.0f / (4.0f * (float)E)));
}

// ---------------------------------------------------------------------------
// Host orchestration
// ---------------------------------------------------------------------------
static inline int grid1(size_t n) { return (int)((n + 255) / 256); }

extern "C" void kernel_launch(void* const* d_in, const int* in_sizes, int n_in,
                              void* d_out, int out_size, void* d_ws, size_t ws_size,
                              hipStream_t stream) {
  (void)in_sizes; (void)n_in; (void)out_size; (void)ws_size;
  const int N = NN, E = EE;

  // ---- inputs (setup_inputs dict order; params flattened in sorted-key order) ----
  const float* x_org     = (const float*)d_in[0];
  const int*   ei        = (const int*)d_in[1];
  const float* edge_attr = (const float*)d_in[2];
  const float* gt_q      = (const float*)d_in[3];
  const float* beta      = (const float*)d_in[4];
  const float* node_feat = (const float*)d_in[5];
  const float* edge_feat = (const float*)d_in[6];
  const int* src = ei;
  const int* dst = ei + E;
  // params base index 7; each _econv -> l1{b,w}, l2{b,w}; _lin -> {b,w};
  // _gat -> {a_dst, a_src, b, w} (sorted keys)
  const int P_CONV1 = 7, P_CONV2 = 11, P_CONV3 = 15, P_SUB1 = 19, P_SUB2 = 23,
            P_SUBPRE = 27, P_SUBSUB = 31, P_CONV4 = 35,
            P_LIN1 = 39, P_LIN2 = 41, P_GAT1 = 43, P_GAT2 = 47;

  float* xfin = (float*)d_out;                 // N x 4
  float* loss = (float*)d_out + (size_t)N * 4; // 1
  float* ores = loss + 1;                      // E x 1

  // ---- workspace carve ----
  char* wptr = (char*)d_ws;
  auto alloc = [&](size_t bytes) -> void* {
    void* p = wptr; wptr += (bytes + 255) & ~(size_t)255; return p;
  };
  float* eam  = (float*)alloc((size_t)E * 4 * 4);
  float* e1   = (float*)alloc((size_t)E * 64 * 4);
  float* e2   = (float*)alloc((size_t)E * 64 * 4);
  float* e3   = (float*)alloc((size_t)E * 64 * 4);
  float* e4   = (float*)alloc((size_t)E * 64 * 4);
  float* e5   = (float*)alloc((size_t)E * 64 * 4);
  float* es1  = (float*)alloc((size_t)E * 128 * 4);
  float* es2  = (float*)alloc((size_t)E * 128 * 4);
  float* ess1 = (float*)alloc((size_t)E * 128 * 4);
  float* etmp = (float*)alloc((size_t)E * 128 * 4);
  float* x1   = (float*)alloc((size_t)N * 64 * 4);
  float* x2   = (float*)alloc((size_t)N * 64 * 4);
  float* x3   = (float*)alloc((size_t)N * 64 * 4);
  float* x4   = (float*)alloc((size_t)N * 64 * 4);
  float* x5   = (float*)alloc((size_t)N * 64 * 4);
  float* xs1  = (float*)alloc((size_t)N * 128 * 4);
  float* xp1  = (float*)alloc((size_t)K1N * 128 * 4);
  float* xs2  = (float*)alloc((size_t)K1N * 128 * 4);
  float* xs2b = (float*)alloc((size_t)K1N * 128 * 4);
  float* xs3  = (float*)alloc((size_t)K1N * 64 * 4);
  float* xp2  = (float*)alloc((size_t)K2N * 128 * 4);
  float* xss1 = (float*)alloc((size_t)K2N * 128 * 4);
  float* xss2 = (float*)alloc((size_t)K2N * 128 * 4);
  float* sbuf = (float*)alloc((size_t)N * 128 * 4);
  float* c0   = (float*)alloc((size_t)N * 4);
  float* c1   = (float*)alloc((size_t)K1N * 4);
  float* c2   = (float*)alloc((size_t)K2N * 4);
  float* hbuf = (float*)alloc((size_t)N * 4);
  float* den  = (float*)alloc((size_t)N * 4);
  float* num  = (float*)alloc((size_t)N * 4);
  float* scr  = (float*)alloc((size_t)N * 4);
  int*   mx   = (int*)alloc((size_t)N * 4);
  int*   mm   = (int*)alloc(2 * 4);
  int*   hist = (int*)alloc(1024 * 4);
  int*   thr  = (int*)alloc(2 * 4);
  int*   ctr  = (int*)alloc(2 * 4);
  int*   perm1 = (int*)alloc((size_t)K1N * 4);
  float* vals1 = (float*)alloc((size_t)K1N * 4);
  int*   remap1 = (int*)alloc((size_t)N * 4);
  int*   perm2 = (int*)alloc((size_t)K2N * 4);
  float* vals2 = (float*)alloc((size_t)K2N * 4);
  int*   remap2 = (int*)alloc((size_t)K1N * 4);
  int*   s1 = (int*)alloc((size_t)E * 4);
  int*   d1 = (int*)alloc((size_t)E * 4);
  float* m1 = (float*)alloc((size_t)E * 4);
  int*   s2 = (int*)alloc((size_t)E * 4);
  int*   d2 = (int*)alloc((size_t)E * 4);
  float* m2 = (float*)alloc((size_t)E * 4);
  _Float16* w1t = (_Float16*)alloc((size_t)128 * 448 * 2);  // max dout x din_p
  _Float16* w2t = (_Float16*)alloc((size_t)128 * 128 * 2);

  const int eb = grid1(E), nb = grid1(N);
  const int econv_blocks = (E + 127) / 128;

  auto run_econv = [&](int dout, const float* xa, int wa, const float* xb, int wb,
                       const float* ea, int wea, const float* ebp, int web,
                       const int* ss, const int* dd, const float* mk,
                       int pidx, float* e_out, float* x_out, const float* cnt, int nnodes) {
    EConvIn A;
    A.xa = xa; A.xb = xb; A.wa = wa; A.wb = wb; A.nin = wa + wb;
    A.ea = ea; A.eb = ebp; A.wea = wea; A.web = web;
    A.src = ss; A.dst = dd; A.mask = mk;
    A.din = 2 * A.nin + wea + web;
    A.din_p = (A.din + 31) & ~31;
    // pre-convert weights: w1 -> N-major f16 K-padded panel; same for w2
    wcvt_kernel<<<grid1((size_t)dout * A.din_p), 256, 0, stream>>>(
        w1t, (const float*)d_in[pidx + 1], A.din, A.din_p, dout);
    wcvt_kernel<<<grid1((size_t)dout * dout), 256, 0, stream>>>(
        w2t, (const float*)d_in[pidx + 3], dout, dout, dout);
    A.w1t = w1t; A.w2t = w2t;
    A.b1 = (const float*)d_in[pidx + 0];
    A.b2 = (const float*)d_in[pidx + 2];
    A.e_out = e_out; A.s_out = sbuf;
    A.E = E;
    (void)hipMemsetAsync(sbuf, 0, (size_t)nnodes * dout * 4, stream);
    if (dout == 64)
      econv_fused<64><<<econv_blocks, 256, 0, stream>>>(A);
    else
      econv_fused<128><<<econv_blocks, 256, 0, stream>>>(A);
    node_fin_kernel<<<grid1((size_t)nnodes * dout), 256, 0, stream>>>(x_out, sbuf, cnt, nnodes, dout);
    relu_kernel<<<grid1((size_t)E * dout), 256, 0, stream>>>(e_out, (size_t)E * dout);
  };

  auto run_gat_pool = [&](const float* x, int nnodes, int k,
                          const int* ss, const int* dd, const float* mk, int pgat,
                          float* xp, int* perm, float* vals, int* remap,
                          int* nsrc, int* ndst, float* nmask) {
    const float* g_adst = (const float*)d_in[pgat + 0];
    const float* g_asrc = (const float*)d_in[pgat + 1];
    const float* g_b    = (const float*)d_in[pgat + 2];
    const float* g_w    = (const float*)d_in[pgat + 3];
    gat_h_kernel<<<grid1(nnodes), 256, 0, stream>>>(hbuf, x, g_w, nnodes, 128);
    fill_int_kernel<<<grid1(nnodes), 256, 0, stream>>>(mx, fkey_host(-FLT_MAX), nnodes);
    gat_max_edges_kernel<<<eb, 256, 0, stream>>>(mx, hbuf, ss, dd, mk, g_asrc, g_adst, E);
    gat_max_self_kernel<<<grid1(nnodes), 256, 0, stream>>>(mx, hbuf, g_asrc, g_adst, nnodes);
    (void)hipMemsetAsync(den, 0, (size_t)nnodes * 4, stream);
    (void)hipMemsetAsync(num, 0, (size_t)nnodes * 4, stream);
    gat_acc_edges_kernel<<<eb, 256, 0, stream>>>(den, num, mx, hbuf, ss, dd, mk, g_asrc, g_adst, E);
    gat_acc_self_kernel<<<grid1(nnodes), 256, 0, stream>>>(den, num, mx, hbuf, g_asrc, g_adst, nnodes);
    gat_score_kernel<<<grid1(nnodes), 256, 0, stream>>>(scr, num, den, g_b, nnodes);
    // top-k threshold selection
    minmax_init_kernel<<<1, 1, 0, stream>>>(mm);
    minmax_kernel<<<grid1(nnodes), 256, 0, stream>>>(mm, scr, nnodes);
    (void)hipMemsetAsync(hist, 0, 1024 * 4, stream);
    hist_kernel<<<grid1(nnodes), 256, 0, stream>>>(hist, mm, scr, nnodes);
    threshold_kernel<<<1, 1, 0, stream>>>(thr, hist, k);
    (void)hipMemsetAsync(ctr, 0, 2 * 4, stream);
    (void)hipMemsetAsync(remap, 0xFF, (size_t)nnodes * 4, stream);
    select_kernel<<<grid1(nnodes), 256, 0, stream>>>(perm, vals, remap, ctr, thr, mm, scr, nnodes, k);
    pool_rows_kernel<<<grid1((size_t)k * 128), 256, 0, stream>>>(xp, x, perm, vals, k, 128);
    remap_edges_kernel<<<eb, 256, 0, stream>>>(nsrc, ndst, nmask, ss, dd, mk, remap, E);
  };

  // ================= forward pass =================
  eam_kernel<<<eb, 256, 0, stream>>>(eam, x_org, edge_attr, src, dst, E);

  (void)hipMemsetAsync(c0, 0, (size_t)N * 4, stream);
  seg_count_kernel<<<eb, 256, 0, stream>>>(c0, dst, nullptr, E);

  // conv1 .. conv3
  run_econv(64, node_feat, 128, x_org, 4, edge_feat, 128, eam, 4,
            src, dst, nullptr, P_CONV1, e1, x1, c0, N);
  run_econv(64, x1, 64, nullptr, 0, eam, 4, e1, 64,
            src, dst, nullptr, P_CONV2, e2, x2, c0, N);
  run_econv(64, x2, 64, x1, 64, e2, 64, e1, 64,
            src, dst, nullptr, P_CONV3, e3, x3, c0, N);
  run_econv(128, x3, 64, nullptr, 0, e3, 64, nullptr, 0,
            src, dst, nullptr, P_SUBPRE, es1, xs1, c0, N);

  // pool1
  run_gat_pool(xs1, N, K1N, src, dst, nullptr, P_GAT1,
               xp1, perm1, vals1, remap1, s1, d1, m1);
  (void)hipMemsetAsync(c1, 0, (size_t)K1N * 4, stream);
  seg_count_kernel<<<eb, 256, 0, stream>>>(c1, d1, m1, E);

  run_econv(128, xp1, 128, nullptr, 0, es1, 128, nullptr, 0,
            s1, d1, m1, P_SUB1, es2, xs2, c1, K1N);

  // pool2
  run_gat_pool(xs2, K1N, K2N, s1, d1, m1, P_GAT2,
               xp2, perm2, vals2, remap2, s2, d2, m2);
  (void)hipMemsetAsync(c2, 0, (size_t)K2N * 4, stream);
  seg_count_kernel<<<eb, 256, 0, stream>>>(c2, d2, m2, E);

  run_econv(128, xp2, 128, nullptr, 0, es2, 128, nullptr, 0,
            s2, d2, m2, P_SUBSUB, ess1, xss1, c2, K2N);
  run_econv(128, xss1, 128, nullptr, 0, ess1, 128, nullptr, 0,
            s2, d2, m2, P_SUBSUB, etmp, xss2, c2, K2N);

  (void)hipMemcpyAsync(xs2b, xs2, (size_t)K1N * 128 * 4, hipMemcpyDeviceToDevice, stream);
  rows_add_kernel<<<grid1((size_t)K2N * 128), 256, 0, stream>>>(xs2b, xss2, perm2, K2N, 128);

  run_econv(64, xs2b, 128, nullptr, 0, es2, 128, nullptr, 0,
            s1, d1, m1, P_SUB2, etmp, xs3, c1, K1N);
  rows_add_kernel<<<grid1((size_t)K1N * 64), 256, 0, stream>>>(x3, xs3, perm1, K1N, 64);

  // conv4 x2
  run_econv(64, x3, 64, x2, 64, e3, 64, e2, 64,
            src, dst, nullptr, P_CONV4, e4, x4, c0, N);
  run_econv(64, x4, 64, x3, 64, e4, 64, e3, 64,
            src, dst, nullptr, P_CONV4, e5, x5, c0, N);

  // heads
  lin1_final_kernel<<<nb, 256, 0, stream>>>(xfin, x5,
      (const float*)d_in[P_LIN1 + 1], (const float*)d_in[P_LIN1 + 0], x_org, N);
  lin2_kernel<<<eb, 256, 0, stream>>>(ores, e5,
      (const float*)d_in[P_LIN2 + 1], (const float*)d_in[P_LIN2 + 0], E);

  (void)hipMemsetAsync(loss, 0, 4, stream);
  loss_kernel<<<eb, 256, 0, stream>>>(loss, xfin, gt_q, src, dst, beta, E);
}
